// TrojanGNN_14714557956359
// MI455X (gfx1250) — compile-verified, matrix operand order checked by reading, not run on hardware
//
#include <hip/hip_runtime.h>
#include <hip/hip_bf16.h>

// ---------------------------------------------------------------------------
// GNN forward for MI455X (gfx1250): bf16 WMMA for all dense transforms,
// async global->LDS weight staging (CDNA5 ASYNCcnt path),
// bf16 message gather + fp32 atomic scatter for the GCN aggregation,
// wave32 shuffle reductions for LayerNorm.
// ---------------------------------------------------------------------------

#define HDIM 128
#define GNUM 64
#define NLAY 4
#define LNEPS 1e-5f

typedef __attribute__((ext_vector_type(16))) __bf16 v16bf;
typedef __attribute__((ext_vector_type(8)))  float  v8f;

__device__ __forceinline__ unsigned short f2bf(float f) {
    unsigned u = __float_as_uint(f);
    unsigned r = u + 0x7FFFu + ((u >> 16) & 1u);   // round-to-nearest-even
    return (unsigned short)(r >> 16);
}
__device__ __forceinline__ float bf2f(unsigned short h) {
    return __uint_as_float(((unsigned)h) << 16);
}

// ------------------------- small utility kernels ---------------------------

__global__ void k_cast_bf16(const float* __restrict__ src,
                            unsigned short* __restrict__ dst, int n) {
    int i = blockIdx.x * blockDim.x + threadIdx.x;
    if (i < n) dst[i] = f2bf(src[i]);
}

__global__ void k_fill_f32(float* __restrict__ p, float v, int n) {
    int i = blockIdx.x * blockDim.x + threadIdx.x;
    if (i < n) p[i] = v;
}

__global__ void k_deg_count(const int* __restrict__ dst, float* __restrict__ deg, int e) {
    int i = blockIdx.x * blockDim.x + threadIdx.x;
    if (i < e) atomicAdd(&deg[dst[i]], 1.0f);
}

__global__ void k_rsqrt_inplace(float* __restrict__ p, int n) {
    int i = blockIdx.x * blockDim.x + threadIdx.x;
    if (i < n) p[i] = rsqrtf(p[i]);
}

// ----------------------- WMMA GEMM:  C = A @ W^T (+b) ----------------------
// A: nrows x K bf16 (row major).  W: HOUT x K bf16 (row major).
// One wave computes a 16-row strip across all HOUT columns.
// Block = 8 waves = 128 rows.  Weights staged in padded LDS via the CDNA5
// asynchronous global->LDS path (no VGPR round trip), tracked by ASYNCcnt.

template <int K, int HOUT>
__global__ __launch_bounds__(256) void
k_gemm_wmma(const unsigned short* __restrict__ Abf,
            const unsigned short* __restrict__ Wbf,
            const float* __restrict__ bias,          // HOUT or null
            float* __restrict__ outF,                // nrows x HOUT or null
            unsigned short* __restrict__ outB,       // nrows x HOUT or null
            int nrows, int do_relu) {
    constexpr int KP = K + 8;        // padded LDS row stride (halves)
    constexpr int NT = HOUT / 16;    // 16-wide column tiles
    constexpr int KC = K / 32;       // 32-deep bf16 K chunks
    __shared__ __align__(16) unsigned short Wl[HOUT * KP];

    const int tid = threadIdx.x;
#if defined(__gfx1250__)
    // Async copy W -> LDS, 16 bytes per lane per issue (GLOBAL_LOAD_ASYNC_TO_LDS_B128).
    // VDST holds the per-lane LDS byte offset; low 32 bits of a generic pointer
    // into __shared__ are the wave-relative LDS address (ISA 10.2 aperture rule).
    for (int idx = tid; idx < HOUT * K / 8; idx += 256) {
        int r = (idx * 8) / K;
        int kk = (idx * 8) % K;
        unsigned lds_off = (unsigned)(unsigned long long)(&Wl[r * KP + kk]);
        unsigned long long gaddr = (unsigned long long)(&Wbf[r * K + kk]);
        asm volatile("global_load_async_to_lds_b128 %0, %1, off"
                     :: "v"(lds_off), "v"(gaddr)
                     : "memory");
    }
    asm volatile("s_wait_asynccnt 0" ::: "memory");
#else
    for (int idx = tid; idx < HOUT * K / 8; idx += 256) {
        int r = (idx * 8) / K;
        int kk = (idx * 8) % K;
        *(uint4*)&Wl[r * KP + kk] = *(const uint4*)&Wbf[r * K + kk];
    }
#endif
    __syncthreads();

    const int lane = tid & 31;
    const int wv   = tid >> 5;
    const int row0 = (blockIdx.x * 8 + wv) * 16;
    const int m    = lane & 15;
    const int hi   = lane >> 4;       // 0 or 1 (half-wave)
    const int arow = row0 + m;
    const bool full_tile = (row0 + 16 <= nrows);

    // A layout (ISA 16-bit A 16x32): lane m holds, per 32-K chunk,
    // halves j=0..7 -> K = base, j=8..15 -> K = base+16, base = kc*32 + 8*hi.
    union AV { uint4 q[2]; v16bf v; };
    AV a[KC];
#pragma unroll
    for (int kc = 0; kc < KC; kc++) {
        int base = kc * 32 + hi * 8;
        if (arow < nrows) {
            a[kc].q[0] = *(const uint4*)&Abf[(size_t)arow * K + base];
            a[kc].q[1] = *(const uint4*)&Abf[(size_t)arow * K + base + 16];
        } else {
            a[kc].q[0] = make_uint4(0u, 0u, 0u, 0u);
            a[kc].q[1] = make_uint4(0u, 0u, 0u, 0u);
        }
    }

#pragma unroll
    for (int ct = 0; ct < NT; ct++) {
        v8f acc = {};
        const int col = ct * 16 + m;     // B column held by this lane
#pragma unroll
        for (int kc = 0; kc < KC; kc++) {
            AV b;
            int base = kc * 32 + hi * 8;
            b.q[0] = *(const uint4*)&Wl[col * KP + base];
            b.q[1] = *(const uint4*)&Wl[col * KP + base + 16];
            acc = __builtin_amdgcn_wmma_f32_16x16x32_bf16(
                false, a[kc].v, false, b.v, (short)0, acc, false, false);
        }
        const float bv = bias ? bias[col] : 0.0f;
        float res[8];
#pragma unroll
        for (int r = 0; r < 8; r++) {
            float v = acc[r] + bv;
            res[r] = do_relu ? fmaxf(v, 0.0f) : v;
        }
        const size_t obase = (size_t)(row0 + hi * 8) * HOUT + col;
        if (full_tile) {
            if (outF) {
#pragma unroll
                for (int r = 0; r < 8; r++) outF[obase + (size_t)r * HOUT] = res[r];
            }
            if (outB) {
#pragma unroll
                for (int r = 0; r < 8; r++) outB[obase + (size_t)r * HOUT] = f2bf(res[r]);
            }
        } else {
#pragma unroll
            for (int r = 0; r < 8; r++) {
                const int orow = row0 + r + hi * 8;
                if (orow < nrows) {
                    const size_t o = (size_t)orow * HOUT + col;
                    if (outF) outF[o] = res[r];
                    if (outB) outB[o] = f2bf(res[r]);
                }
            }
        }
    }
}

// -------------------- scatter init: self-loop + conv bias ------------------
// ACC[n][c] = convB[c] + h[n][c] * dinv[n]^2      (32 threads per node)

__global__ __launch_bounds__(256) void
k_acc_init(const unsigned short* __restrict__ Hbf, const float* __restrict__ dinv,
           const float* __restrict__ bias, float* __restrict__ acc, int n) {
    int t = blockIdx.x * blockDim.x + threadIdx.x;
    int node = t >> 5;
    if (node >= n) return;
    int c = (t & 31) * 4;
    float dv = dinv[node];
    float s = dv * dv;
    const unsigned short* hp = &Hbf[(size_t)node * HDIM + c];
    float* ap = &acc[(size_t)node * HDIM + c];
    uint2 q = *(const uint2*)hp;
#pragma unroll
    for (int j = 0; j < 4; j++) {
        unsigned short hv = (unsigned short)((j < 2 ? q.x : q.y) >> ((j & 1) * 16));
        ap[j] = bias[c + j] + bf2f(hv) * s;
    }
}

// ------------------ edge scatter: ACC[dst] += h[src] * norm ----------------
// One wave per edge: coalesced bf16 gather of the full 128-wide row,
// fp32 GLOBAL_ATOMIC_ADD per element.

__global__ __launch_bounds__(256) void
k_scatter(const int* __restrict__ src, const int* __restrict__ dst,
          const unsigned short* __restrict__ Hbf, const float* __restrict__ dinv,
          float* __restrict__ acc, int e) {
    int w = (blockIdx.x * blockDim.x + threadIdx.x) >> 5;
    if (w >= e) return;
    int lane = threadIdx.x & 31;
    int s = src[w];
    int d = dst[w];
    float nm = dinv[s] * dinv[d];
    const unsigned short* hp = &Hbf[(size_t)s * HDIM + lane * 4];
    __builtin_prefetch(hp, 0, 1);
    float* ap = &acc[(size_t)d * HDIM + lane * 4];
    uint2 q = *(const uint2*)hp;
    atomicAdd(&ap[0], bf2f((unsigned short)(q.x & 0xffffu)) * nm);
    atomicAdd(&ap[1], bf2f((unsigned short)(q.x >> 16)) * nm);
    atomicAdd(&ap[2], bf2f((unsigned short)(q.y & 0xffffu)) * nm);
    atomicAdd(&ap[3], bf2f((unsigned short)(q.y >> 16)) * nm);
}

// ------------- LayerNorm + ReLU (+ optional residual), wave/node -----------

__global__ __launch_bounds__(256) void
k_ln_relu(const float* __restrict__ in, const float* __restrict__ gam,
          const float* __restrict__ bet, float* __restrict__ Xres,  // in/out or null
          unsigned short* __restrict__ outB, int n) {
    int w = (blockIdx.x * blockDim.x + threadIdx.x) >> 5;
    if (w >= n) return;
    int lane = threadIdx.x & 31;
    const float* ip = &in[(size_t)w * HDIM + lane * 4];
    float4 v = *(const float4*)ip;

    float s = v.x + v.y + v.z + v.w;
#pragma unroll
    for (int o = 16; o > 0; o >>= 1) s += __shfl_xor(s, o, 32);
    float mean = s * (1.0f / HDIM);

    float dx = v.x - mean, dy = v.y - mean, dz = v.z - mean, dw = v.w - mean;
    float q = dx * dx + dy * dy + dz * dz + dw * dw;
#pragma unroll
    for (int o = 16; o > 0; o >>= 1) q += __shfl_xor(q, o, 32);
    float inv = rsqrtf(q * (1.0f / HDIM) + LNEPS);

    int c = lane * 4;
    float y0 = fmaxf(dx * inv * gam[c + 0] + bet[c + 0], 0.0f);
    float y1 = fmaxf(dy * inv * gam[c + 1] + bet[c + 1], 0.0f);
    float y2 = fmaxf(dz * inv * gam[c + 2] + bet[c + 2], 0.0f);
    float y3 = fmaxf(dw * inv * gam[c + 3] + bet[c + 3], 0.0f);

    if (Xres) {
        float4 xo = *(const float4*)&Xres[(size_t)w * HDIM + c];
        y0 += xo.x; y1 += xo.y; y2 += xo.z; y3 += xo.w;
        *(float4*)&Xres[(size_t)w * HDIM + c] = make_float4(y0, y1, y2, y3);
    }
    if (outB) {
        uint2 pk;
        pk.x = (unsigned)f2bf(y0) | ((unsigned)f2bf(y1) << 16);
        pk.y = (unsigned)f2bf(y2) | ((unsigned)f2bf(y3) << 16);
        *(uint2*)&outB[(size_t)w * HDIM + c] = pk;
    }
}

// ------------------------------ pooling ------------------------------------

__global__ void k_pool_init(float* __restrict__ gsum, float* __restrict__ gmax,
                            float* __restrict__ gcnt) {
    int i = blockIdx.x * blockDim.x + threadIdx.x;
    if (i < GNUM * HDIM) { gsum[i] = 0.0f; gmax[i] = -3.402823466e38f; }
    if (i < GNUM) gcnt[i] = 0.0f;
}

__device__ __forceinline__ void atomicMaxF(float* addr, float v) {
    if (v >= 0.0f) atomicMax((int*)addr, __float_as_int(v));
    else           atomicMin((unsigned*)addr, __float_as_uint(v));
}

__global__ __launch_bounds__(256) void
k_pool(const float* __restrict__ X, const int* __restrict__ batch,
       float* __restrict__ gsum, float* __restrict__ gmax,
       float* __restrict__ gcnt, int n) {
    int w = (blockIdx.x * blockDim.x + threadIdx.x) >> 5;
    if (w >= n) return;
    int lane = threadIdx.x & 31;
    int b = batch[w];
    int c = lane * 4;
    float4 v = *(const float4*)&X[(size_t)w * HDIM + c];
    atomicAdd(&gsum[b * HDIM + c + 0], v.x);
    atomicAdd(&gsum[b * HDIM + c + 1], v.y);
    atomicAdd(&gsum[b * HDIM + c + 2], v.z);
    atomicAdd(&gsum[b * HDIM + c + 3], v.w);
    atomicMaxF(&gmax[b * HDIM + c + 0], v.x);
    atomicMaxF(&gmax[b * HDIM + c + 1], v.y);
    atomicMaxF(&gmax[b * HDIM + c + 2], v.z);
    atomicMaxF(&gmax[b * HDIM + c + 3], v.w);
    if (lane == 0) atomicAdd(&gcnt[b], 1.0f);
}

// ----------------------------- graph head ----------------------------------
// One block of 128 threads per graph; everything here is tiny (64 graphs).

__global__ __launch_bounds__(128) void
k_graph_head(const float* __restrict__ gsum, const float* __restrict__ gmax,
             const float* __restrict__ gcnt,
             const float* __restrict__ W1, const float* __restrict__ b1,
             const float* __restrict__ gg, const float* __restrict__ gb,
             const float* __restrict__ W2, const float* __restrict__ b2,
             const float* __restrict__ W3, const float* __restrict__ b3,
             float* __restrict__ out) {
    __shared__ float emb[2 * HDIM];
    __shared__ float h1s[HDIM];
    __shared__ float h2s[HDIM / 2];
    __shared__ float red[4];
    int t = threadIdx.x;
    int g = blockIdx.x;

    float cnt = fmaxf(gcnt[g], 1.0f);
    emb[t] = gsum[g * HDIM + t] / cnt;
    emb[HDIM + t] = gmax[g * HDIM + t];
    __syncthreads();

    float a = b1[t];
    for (int k = 0; k < 2 * HDIM; k++) a = fmaf(W1[t * 2 * HDIM + k], emb[k], a);

    float s = a;
#pragma unroll
    for (int o = 16; o > 0; o >>= 1) s += __shfl_xor(s, o, 32);
    if ((t & 31) == 0) red[t >> 5] = s;
    __syncthreads();
    float mean = (red[0] + red[1] + red[2] + red[3]) * (1.0f / HDIM);
    __syncthreads();

    float d = a - mean;
    float q = d * d;
#pragma unroll
    for (int o = 16; o > 0; o >>= 1) q += __shfl_xor(q, o, 32);
    if ((t & 31) == 0) red[t >> 5] = q;
    __syncthreads();
    float var = (red[0] + red[1] + red[2] + red[3]) * (1.0f / HDIM);

    h1s[t] = fmaxf(d * rsqrtf(var + LNEPS) * gg[t] + gb[t], 0.0f);
    __syncthreads();

    if (t < HDIM / 2) {
        float a2 = b2[t];
        for (int k = 0; k < HDIM; k++) a2 = fmaf(W2[t * HDIM + k], h1s[k], a2);
        h2s[t] = fmaxf(a2, 0.0f);
    }
    __syncthreads();

    if (t < 2) {
        float a3 = b3[t];
        for (int k = 0; k < HDIM / 2; k++) a3 = fmaf(W3[t * (HDIM / 2) + k], h2s[k], a3);
        out[g * 2 + t] = a3;
    }
}

// ----------------------------- node logits ---------------------------------

__global__ void k_node_logits(const float* __restrict__ nh2,
                              const float* __restrict__ W3,
                              const float* __restrict__ b3,
                              float* __restrict__ out, int n) {
    int i = blockIdx.x * blockDim.x + threadIdx.x;
    if (i >= n) return;
    const float* p = &nh2[(size_t)i * (HDIM / 2)];
    float a0 = b3[0], a1 = b3[1];
    for (int k = 0; k < HDIM / 2; k++) {
        float v = p[k];
        a0 = fmaf(W3[k], v, a0);
        a1 = fmaf(W3[HDIM / 2 + k], v, a1);
    }
    out[(size_t)i * 2 + 0] = a0;
    out[(size_t)i * 2 + 1] = a1;
}

// ---------------------------------------------------------------------------

extern "C" void kernel_launch(void* const* d_in, const int* in_sizes, int n_in,
                              void* d_out, int out_size, void* d_ws, size_t ws_size,
                              hipStream_t stream) {
    const float* x      = (const float*)d_in[0];
    const int*   ei     = (const int*)d_in[1];
    const int*   batch  = (const int*)d_in[2];
    const float* W_in   = (const float*)d_in[3];
    const float* b_in   = (const float*)d_in[4];
    const float* convW  = (const float*)d_in[5];
    const float* convB  = (const float*)d_in[6];
    const float* lnG    = (const float*)d_in[7];
    const float* lnB    = (const float*)d_in[8];
    const float* gh_W1  = (const float*)d_in[9];
    const float* gh_b1  = (const float*)d_in[10];
    const float* gh_g   = (const float*)d_in[11];
    const float* gh_be  = (const float*)d_in[12];
    const float* gh_W2  = (const float*)d_in[13];
    const float* gh_b2  = (const float*)d_in[14];
    const float* gh_W3  = (const float*)d_in[15];
    const float* gh_b3  = (const float*)d_in[16];
    const float* nh_W1  = (const float*)d_in[17];
    const float* nh_b1  = (const float*)d_in[18];
    const float* nh_g   = (const float*)d_in[19];
    const float* nh_be  = (const float*)d_in[20];
    const float* nh_W2  = (const float*)d_in[21];
    const float* nh_b2  = (const float*)d_in[22];
    const float* nh_W3  = (const float*)d_in[23];
    const float* nh_b3  = (const float*)d_in[24];

    const int N = in_sizes[0] / HDIM;
    const int E = in_sizes[1] / 2;
    const int* e_src = ei;
    const int* e_dst = ei + E;

    // ----- workspace carve-out -----
    char* w = (char*)d_ws;
    size_t off = 0;
    auto carve = [&](size_t bytes) -> void* {
        void* p = w + off;
        off = (off + bytes + 255) & ~(size_t)255;
        return p;
    };
    unsigned short* A_bf    = (unsigned short*)carve((size_t)N * HDIM * 2);
    unsigned short* H_bf    = (unsigned short*)carve((size_t)N * HDIM * 2);
    float*          Xf      = (float*)carve((size_t)N * HDIM * 4);
    float*          ACC     = (float*)carve((size_t)N * HDIM * 4);
    float*          NH2     = (float*)carve((size_t)N * (HDIM / 2) * 4);
    float*          dinv    = (float*)carve((size_t)N * 4);
    unsigned short* wbf_in  = (unsigned short*)carve((size_t)HDIM * HDIM * 2);
    unsigned short* wbf_cv  = (unsigned short*)carve((size_t)NLAY * HDIM * HDIM * 2);
    unsigned short* wbf_n1  = (unsigned short*)carve((size_t)HDIM * HDIM * 2);
    unsigned short* wbf_n2  = (unsigned short*)carve((size_t)(HDIM / 2) * HDIM * 2);
    float*          gsum    = (float*)carve((size_t)GNUM * HDIM * 4);
    float*          gmax    = (float*)carve((size_t)GNUM * HDIM * 4);
    float*          gcnt    = (float*)carve((size_t)GNUM * 4);
    (void)ws_size; (void)n_in; (void)out_size;

    const int T = 256;
    auto cdiv = [](long long a, long long b) { return (int)((a + b - 1) / b); };

    // weights + activations to bf16
    k_cast_bf16<<<cdiv(HDIM * HDIM, T), T, 0, stream>>>(W_in, wbf_in, HDIM * HDIM);
    k_cast_bf16<<<cdiv(NLAY * HDIM * HDIM, T), T, 0, stream>>>(convW, wbf_cv, NLAY * HDIM * HDIM);
    k_cast_bf16<<<cdiv(HDIM * HDIM, T), T, 0, stream>>>(nh_W1, wbf_n1, HDIM * HDIM);
    k_cast_bf16<<<cdiv((HDIM / 2) * HDIM, T), T, 0, stream>>>(nh_W2, wbf_n2, (HDIM / 2) * HDIM);
    k_cast_bf16<<<cdiv((long long)N * HDIM, T), T, 0, stream>>>(x, A_bf, N * HDIM);

    // GCN degree normalization (deg includes self-loop)
    k_fill_f32<<<cdiv(N, T), T, 0, stream>>>(dinv, 1.0f, N);
    k_deg_count<<<cdiv(E, T), T, 0, stream>>>(e_dst, dinv, E);
    k_rsqrt_inplace<<<cdiv(N, T), T, 0, stream>>>(dinv, N);

    const int gemm_blocks = cdiv(N, 128);

    // input projection + ReLU -> Xf (fp32 features) and A_bf (bf16 mirror)
    k_gemm_wmma<HDIM, HDIM><<<gemm_blocks, T, 0, stream>>>(
        A_bf, wbf_in, b_in, Xf, A_bf, N, 1);

    // 4 GCN layers
    for (int i = 0; i < NLAY; i++) {
        k_gemm_wmma<HDIM, HDIM><<<gemm_blocks, T, 0, stream>>>(
            A_bf, wbf_cv + (size_t)i * HDIM * HDIM, nullptr, nullptr, H_bf, N, 0);
        k_acc_init<<<cdiv((long long)N * 32, T), T, 0, stream>>>(
            H_bf, dinv, convB + i * HDIM, ACC, N);
        k_scatter<<<cdiv((long long)E * 32, T), T, 0, stream>>>(
            e_src, e_dst, H_bf, dinv, ACC, E);
        k_ln_relu<<<cdiv((long long)N * 32, T), T, 0, stream>>>(
            ACC, lnG + i * HDIM, lnB + i * HDIM, Xf, A_bf, N);
    }

    // pooling over graphs
    k_pool_init<<<cdiv(GNUM * HDIM, T), T, 0, stream>>>(gsum, gmax, gcnt);
    k_pool<<<cdiv((long long)N * 32, T), T, 0, stream>>>(Xf, batch, gsum, gmax, gcnt, N);

    // graph head -> d_out[0 .. 2G)
    k_graph_head<<<GNUM, 128, 0, stream>>>(gsum, gmax, gcnt,
                                           gh_W1, gh_b1, gh_g, gh_be,
                                           gh_W2, gh_b2, gh_W3, gh_b3,
                                           (float*)d_out);

    // node head: lin -> LN+ReLU -> lin+ReLU -> logits
    k_gemm_wmma<HDIM, HDIM><<<gemm_blocks, T, 0, stream>>>(
        A_bf, wbf_n1, nh_b1, ACC, nullptr, N, 0);
    k_ln_relu<<<cdiv((long long)N * 32, T), T, 0, stream>>>(
        ACC, nh_g, nh_be, nullptr, A_bf, N);
    k_gemm_wmma<HDIM, HDIM / 2><<<gemm_blocks, T, 0, stream>>>(
        A_bf, wbf_n2, nh_b2, NH2, nullptr, N, 1);
    k_node_logits<<<cdiv(N, T), T, 0, stream>>>(
        NH2, nh_W3, nh_b3, (float*)d_out + 2 * GNUM, N);
}
